// NestedTensorBlock_63118839382234
// MI455X (gfx1250) — compile-verified, matrix operand order checked by reading, not run on hardware
//
#include <hip/hip_runtime.h>
#include <hip/hip_bf16.h>
#include <stdint.h>

// ---------------------------------------------------------------------------
// ViT block (DINOv2 NestedTensorBlock eval path) for MI455X / gfx1250.
// All matmuls via V_WMMA_F32_16X16X32_BF16 (wave32).
// GEMMs: 2x4 register-blocked tiles per wave (32M x 64N) -> ~21 FLOP/byte.
// ---------------------------------------------------------------------------

typedef __attribute__((ext_vector_type(16))) __bf16 v16bf;
typedef __attribute__((ext_vector_type(8)))  float  v8f;

#define DIMC   1024
#define NHEAD  16
#define DHEAD  64
#define HIDC   4096
#define TTOT   7536            // 4*1370 + 8*257
#define TSPLIT 5480            // tokens of group 0
#define NG0    1370
#define NG1    257
#define NS1    258             // padded (even) row stride for group-1 V^T
#define VT1OFF (4ull*16*64*1370)
#define QT0    86              // ceil(1370/16)
#define QT1    17              // ceil(257/16)
#define BIGNEG (-3.0e38f)
#define MTILES 236             // ceil(7536/32)

union Frag {
  v16bf v;
  unsigned u[8];
  uint4  q4[2];
};

__device__ __forceinline__ unsigned short f2bf(float x) {
  unsigned u = __float_as_uint(x);
  unsigned r = (u + 0x7fffu + ((u >> 16) & 1u)) >> 16;   // round-to-nearest-even
  return (unsigned short)r;
}
__device__ __forceinline__ int imin(int a, int b) { return a < b ? a : b; }

#define WMMA_BF16(A, B, C) \
  __builtin_amdgcn_wmma_f32_16x16x32_bf16(false, (A), false, (B), (short)0, (C), false, false)

// --------------------------- weight f32 -> bf16 transpose -------------------
__global__ void k_wt(const float* __restrict__ W, unsigned short* __restrict__ out,
                     int K, int N) {
  int i = blockIdx.x * 256 + threadIdx.x;            // i indexes out[n*K + k]
  if (i >= K * N) return;
  int n = i / K, k = i - n * K;
  out[i] = f2bf(W[(size_t)k * N + n]);
}

// --------------------------- LayerNorm (wave per token) ---------------------
__global__ void k_ln(const float* __restrict__ xa, const float* __restrict__ xb,
                     int split, const float* __restrict__ g, const float* __restrict__ b,
                     unsigned short* __restrict__ out) {
  int w = threadIdx.x >> 5, lane = threadIdx.x & 31;
  int tok = blockIdx.x * 8 + w;
  const float* x = (tok < split) ? (xa + (size_t)tok * DIMC)
                                 : (xb + (size_t)(tok - split) * DIMC);
  float s = 0.f, ss = 0.f, vals[32];
#pragma unroll
  for (int i = 0; i < 32; ++i) {
    float t = x[lane + 32 * i];
    vals[i] = t; s += t; ss += t * t;
  }
#pragma unroll
  for (int o = 16; o > 0; o >>= 1) { s += __shfl_xor(s, o); ss += __shfl_xor(ss, o); }
  float mean = s * (1.f / DIMC);
  float var  = ss * (1.f / DIMC) - mean * mean;
  float rstd = rsqrtf(var + 1e-5f);
  unsigned short* orow = out + (size_t)tok * DIMC;
#pragma unroll
  for (int i = 0; i < 32; ++i) {
    int c = lane + 32 * i;
    orow[c] = f2bf((vals[i] - mean) * rstd * g[c] + b[c]);
  }
}

// ------------------ 2x4 register-blocked wave GEMM (32M x 64N) --------------
// A: [M,K] bf16 row-major (rows clamped to M-1), Bt: [N,K] bf16 row-major.
__device__ __forceinline__ void gemm_2x4(const unsigned short* __restrict__ A,
                                         const unsigned short* __restrict__ Bt,
                                         int K, int M, int m0, int n0,
                                         v8f c[2][4]) {
  int lane = threadIdx.x & 31;
  int hl = lane >> 4, idx = lane & 15;
  const unsigned short* arow0 = A + (size_t)imin(m0 + idx, M - 1) * K + 8 * hl;
  const unsigned short* arow1 = A + (size_t)imin(m0 + 16 + idx, M - 1) * K + 8 * hl;
  const unsigned short* brow[4];
#pragma unroll
  for (int ni = 0; ni < 4; ++ni)
    brow[ni] = Bt + (size_t)(n0 + ni * 16 + idx) * K + 8 * hl;
#pragma unroll
  for (int mi = 0; mi < 2; ++mi)
#pragma unroll
    for (int ni = 0; ni < 4; ++ni)
#pragma unroll
      for (int v = 0; v < 8; ++v) c[mi][ni][v] = 0.f;

  for (int k0 = 0; k0 < K; k0 += 32) {
    Frag a0, a1, b[4];
    a0.q4[0] = *(const uint4*)(arow0 + k0);
    a0.q4[1] = *(const uint4*)(arow0 + k0 + 16);
    a1.q4[0] = *(const uint4*)(arow1 + k0);
    a1.q4[1] = *(const uint4*)(arow1 + k0 + 16);
#pragma unroll
    for (int ni = 0; ni < 4; ++ni) {
      b[ni].q4[0] = *(const uint4*)(brow[ni] + k0);
      b[ni].q4[1] = *(const uint4*)(brow[ni] + k0 + 16);
    }
#pragma unroll
    for (int ni = 0; ni < 4; ++ni) {
      c[0][ni] = WMMA_BF16(a0.v, b[ni].v, c[0][ni]);
      c[1][ni] = WMMA_BF16(a1.v, b[ni].v, c[1][ni]);
    }
  }
}

// --------------------------- QKV GEMM + split/repack ------------------------
__global__ void k_gemm_qkv(const unsigned short* __restrict__ lnx,
                           const unsigned short* __restrict__ Wt,
                           const float* __restrict__ bias,
                           unsigned short* __restrict__ q,
                           unsigned short* __restrict__ kk,
                           unsigned short* __restrict__ vt) {
  int w = threadIdx.x >> 5;
  int m0 = blockIdx.x * 32;
  int n0 = (blockIdx.y * 8 + w) * 64;
  v8f c[2][4];
  gemm_2x4(lnx, Wt, DIMC, TTOT, m0, n0, c);
  int lane = threadIdx.x & 31, hl = lane >> 4, idx = lane & 15;
#pragma unroll
  for (int ni = 0; ni < 4; ++ni) {
    int col = n0 + ni * 16 + idx;
    float bv = bias[col];
    int s3 = col >> 10, cc = col & 1023;
#pragma unroll
    for (int mi = 0; mi < 2; ++mi) {
#pragma unroll
      for (int v = 0; v < 8; ++v) {
        int tok = m0 + mi * 16 + 8 * hl + v;
        if (tok >= TTOT) continue;
        float val = c[mi][ni][v] + bv;
        if (s3 == 0) {
          q[(size_t)tok * DIMC + cc] = f2bf(val * 0.125f);      // dh^-0.5
        } else if (s3 == 1) {
          kk[(size_t)tok * DIMC + cc] = f2bf(val);
        } else {
          int h = cc >> 6, d = cc & 63;
          size_t addr;
          if (tok < TSPLIT) {
            int s = tok / NG0, pos = tok - s * NG0;
            addr = ((size_t)(s * 16 + h) * 64 + d) * NG0 + pos;
          } else {
            int t2 = tok - TSPLIT;
            int s = t2 / NG1, pos = t2 - s * NG1;
            addr = VT1OFF + ((size_t)(s * 16 + h) * 64 + d) * NS1 + pos;
          }
          vt[addr] = f2bf(val);
        }
      }
    }
  }
}

// --------------------------- flash attention (wave per q-tile) --------------
__global__ void k_flash(const unsigned short* __restrict__ qm,
                        const unsigned short* __restrict__ km,
                        const unsigned short* __restrict__ vt,
                        unsigned short* __restrict__ ao) {
  int w = threadIdx.x >> 5;
  int flat = blockIdx.x * 8 + w;
  int lane = threadIdx.x & 31, hl = lane >> 4, idx = lane & 15;

  int s, h, qt, base, n, ns; size_t vtoff;
  if (flat < 4 * 16 * QT0) {
    qt = flat % QT0; int r = flat / QT0; h = r & 15; s = r >> 4;
    base = s * NG0; n = NG0; ns = NG0; vtoff = 0;
  } else {
    int f2 = flat - 4 * 16 * QT0;
    qt = f2 % QT1; int r = f2 / QT1; h = r & 15; s = r >> 4;
    base = TSPLIT + s * NG1; n = NG1; ns = NS1; vtoff = VT1OFF;
  }

  // Q^T B-fragments (two 32-deep chunks over dh=64). Lane col = query idx.
  int qtok = base + imin(qt * 16 + idx, n - 1);
  const unsigned short* qrow = qm + (size_t)qtok * DIMC + h * DHEAD + 8 * hl;
  Frag bq0, bq1;
  bq0.q4[0] = *(const uint4*)(qrow);
  bq0.q4[1] = *(const uint4*)(qrow + 16);
  bq1.q4[0] = *(const uint4*)(qrow + 32);
  bq1.q4[1] = *(const uint4*)(qrow + 48);

  float mrun = BIGNEG, lrun = 0.f;
  v8f acc[4];
#pragma unroll
  for (int t = 0; t < 4; ++t)
#pragma unroll
    for (int v = 0; v < 8; ++v) acc[t][v] = 0.f;

  const unsigned short* vbase = vt + vtoff;
  int nkt = (n + 31) >> 5;

  for (int kt = 0; kt < nkt; ++kt) {
    int kb = kt * 32;
    v8f st0 = {0.f,0.f,0.f,0.f,0.f,0.f,0.f,0.f};
    v8f st1 = {0.f,0.f,0.f,0.f,0.f,0.f,0.f,0.f};

    { // S^T keys kb..kb+15 : A = K-tile (rows = keys, K-dim = d)
      int ktok = base + imin(kb + idx, n - 1);
      const unsigned short* krow = km + (size_t)ktok * DIMC + h * DHEAD + 8 * hl;
      Frag a;
      a.q4[0] = *(const uint4*)(krow);      a.q4[1] = *(const uint4*)(krow + 16);
      st0 = WMMA_BF16(a.v, bq0.v, st0);
      a.q4[0] = *(const uint4*)(krow + 32); a.q4[1] = *(const uint4*)(krow + 48);
      st0 = WMMA_BF16(a.v, bq1.v, st0);
    }
    { // S^T keys kb+16..kb+31
      int ktok = base + imin(kb + 16 + idx, n - 1);
      const unsigned short* krow = km + (size_t)ktok * DIMC + h * DHEAD + 8 * hl;
      Frag a;
      a.q4[0] = *(const uint4*)(krow);      a.q4[1] = *(const uint4*)(krow + 16);
      st1 = WMMA_BF16(a.v, bq0.v, st1);
      a.q4[0] = *(const uint4*)(krow + 32); a.q4[1] = *(const uint4*)(krow + 48);
      st1 = WMMA_BF16(a.v, bq1.v, st1);
    }

    // mask invalid keys, online softmax over key dim (= fragment rows)
    float s0v[8], s1v[8], mt = BIGNEG;
#pragma unroll
    for (int v = 0; v < 8; ++v) {
      int key0 = kb + 8 * hl + v;
      int key1 = kb + 16 + 8 * hl + v;
      s0v[v] = (key0 < n) ? st0[v] : BIGNEG;
      s1v[v] = (key1 < n) ? st1[v] : BIGNEG;
      mt = fmaxf(mt, fmaxf(s0v[v], s1v[v]));
    }
    mt = fmaxf(mt, __shfl_xor(mt, 16));
    float mnew = fmaxf(mrun, mt);
    float scale = __expf(mrun - mnew);
    float p0[8], p1[8], lsum = 0.f;
#pragma unroll
    for (int v = 0; v < 8; ++v) {
      p0[v] = __expf(s0v[v] - mnew);
      p1[v] = __expf(s1v[v] - mnew);
      lsum += p0[v] + p1[v];
    }
    lsum += __shfl_xor(lsum, 16);
    lrun = lrun * scale + lsum;
    mrun = mnew;
#pragma unroll
    for (int t = 0; t < 4; ++t)
#pragma unroll
      for (int v = 0; v < 8; ++v) acc[t][v] *= scale;

    // P^T (C-layout) -> B-layout is lane-local: pack pairs along keys.
    Frag bp;
#pragma unroll
    for (int j = 0; j < 4; ++j)
      bp.u[j] = (unsigned)f2bf(p0[2 * j]) | ((unsigned)f2bf(p0[2 * j + 1]) << 16);
#pragma unroll
    for (int j = 0; j < 4; ++j)
      bp.u[4 + j] = (unsigned)f2bf(p1[2 * j]) | ((unsigned)f2bf(p1[2 * j + 1]) << 16);

    // O^T += V^T-tile @ P^T  (4 d-tiles of 16)
#pragma unroll
    for (int t = 0; t < 4; ++t) {
      const unsigned* vr32 = (const unsigned*)
          (vbase + ((size_t)((s * 16 + h) * 64 + t * 16 + idx)) * ns);
      Frag av;
      int d0 = (kb + 8 * hl) >> 1, d1 = (kb + 16 + 8 * hl) >> 1;
#pragma unroll
      for (int j = 0; j < 4; ++j) { av.u[j] = vr32[d0 + j]; av.u[4 + j] = vr32[d1 + j]; }
      acc[t] = WMMA_BF16(av.v, bp.v, acc[t]);
    }
  }

  int qg = qt * 16 + idx;
  if (qg < n) {
    float inv = 1.f / lrun;
    unsigned* orow = (unsigned*)(ao + (size_t)(base + qg) * DIMC + h * DHEAD);
#pragma unroll
    for (int t = 0; t < 4; ++t) {
      int dbase = (t * 16 + 8 * hl) >> 1;
#pragma unroll
      for (int j = 0; j < 4; ++j) {
        unsigned pk = (unsigned)f2bf(acc[t][2 * j] * inv) |
                      ((unsigned)f2bf(acc[t][2 * j + 1] * inv) << 16);
        orow[dbase + j] = pk;
      }
    }
  }
}

// --------------------------- proj GEMM + ls1 residual -----------------------
__global__ void k_gemm_proj(const unsigned short* __restrict__ ao,
                            const unsigned short* __restrict__ Wt,
                            const float* __restrict__ bias, const float* __restrict__ ls1,
                            const float* __restrict__ x0, const float* __restrict__ x1,
                            float* __restrict__ hbuf) {
  int w = threadIdx.x >> 5;
  int m0 = blockIdx.x * 32, n0 = (blockIdx.y * 8 + w) * 64;
  v8f c[2][4];
  gemm_2x4(ao, Wt, DIMC, TTOT, m0, n0, c);
  int lane = threadIdx.x & 31, hl = lane >> 4, idx = lane & 15;
#pragma unroll
  for (int ni = 0; ni < 4; ++ni) {
    int col = n0 + ni * 16 + idx;
    float bv = bias[col], lv = ls1[col];
#pragma unroll
    for (int mi = 0; mi < 2; ++mi) {
#pragma unroll
      for (int v = 0; v < 8; ++v) {
        int tok = m0 + mi * 16 + 8 * hl + v;
        if (tok >= TTOT) continue;
        float xin = (tok < TSPLIT) ? x0[(size_t)tok * DIMC + col]
                                   : x1[(size_t)(tok - TSPLIT) * DIMC + col];
        hbuf[(size_t)tok * DIMC + col] = xin + lv * (c[mi][ni][v] + bv);
      }
    }
  }
}

// --------------------------- fc1 GEMM + exact GELU --------------------------
__global__ void k_gemm_fc1(const unsigned short* __restrict__ lnx,
                           const unsigned short* __restrict__ Wt,
                           const float* __restrict__ bias,
                           unsigned short* __restrict__ act) {
  int w = threadIdx.x >> 5;
  int m0 = blockIdx.x * 32, n0 = (blockIdx.y * 8 + w) * 64;
  v8f c[2][4];
  gemm_2x4(lnx, Wt, DIMC, TTOT, m0, n0, c);
  int lane = threadIdx.x & 31, hl = lane >> 4, idx = lane & 15;
#pragma unroll
  for (int ni = 0; ni < 4; ++ni) {
    int col = n0 + ni * 16 + idx;
    float bv = bias[col];
#pragma unroll
    for (int mi = 0; mi < 2; ++mi) {
#pragma unroll
      for (int v = 0; v < 8; ++v) {
        int tok = m0 + mi * 16 + 8 * hl + v;
        if (tok >= TTOT) continue;
        float t = c[mi][ni][v] + bv;
        float g = 0.5f * t * (1.f + erff(t * 0.70710678118654752f));
        act[(size_t)tok * HIDC + col] = f2bf(g);
      }
    }
  }
}

// --------------------------- fc2 GEMM + ls2 residual -> out -----------------
__global__ void k_gemm_fc2(const unsigned short* __restrict__ act,
                           const unsigned short* __restrict__ Wt,
                           const float* __restrict__ bias, const float* __restrict__ ls2,
                           const float* __restrict__ hbuf, float* __restrict__ out) {
  int w = threadIdx.x >> 5;
  int m0 = blockIdx.x * 32, n0 = (blockIdx.y * 8 + w) * 64;
  v8f c[2][4];
  gemm_2x4(act, Wt, HIDC, TTOT, m0, n0, c);
  int lane = threadIdx.x & 31, hl = lane >> 4, idx = lane & 15;
#pragma unroll
  for (int ni = 0; ni < 4; ++ni) {
    int col = n0 + ni * 16 + idx;
    float bv = bias[col], lv = ls2[col];
#pragma unroll
    for (int mi = 0; mi < 2; ++mi) {
#pragma unroll
      for (int v = 0; v < 8; ++v) {
        int tok = m0 + mi * 16 + 8 * hl + v;
        if (tok >= TTOT) continue;
        out[(size_t)tok * DIMC + col] = hbuf[(size_t)tok * DIMC + col] +
                                        lv * (c[mi][ni][v] + bv);
      }
    }
  }
}

// ---------------------------------------------------------------------------
extern "C" void kernel_launch(void* const* d_in, const int* in_sizes, int n_in,
                              void* d_out, int out_size, void* d_ws, size_t ws_size,
                              hipStream_t stream) {
  (void)in_sizes; (void)n_in; (void)out_size; (void)ws_size;
  const float* x0     = (const float*)d_in[0];
  const float* x1     = (const float*)d_in[1];
  const float* ln1_g  = (const float*)d_in[2];
  const float* ln1_b  = (const float*)d_in[3];
  const float* qkv_w  = (const float*)d_in[4];
  const float* qkv_b  = (const float*)d_in[5];
  const float* proj_w = (const float*)d_in[6];
  const float* proj_b = (const float*)d_in[7];
  const float* ls1    = (const float*)d_in[8];
  const float* ln2_g  = (const float*)d_in[9];
  const float* ln2_b  = (const float*)d_in[10];
  const float* fc1_w  = (const float*)d_in[11];
  const float* fc1_b  = (const float*)d_in[12];
  const float* fc2_w  = (const float*)d_in[13];
  const float* fc2_b  = (const float*)d_in[14];
  const float* ls2    = (const float*)d_in[15];

  char* ws = (char*)d_ws;
  size_t off = 0;
  auto alloc = [&](size_t bytes) -> size_t {
    size_t o = off; off += (bytes + 255) & ~(size_t)255; return o;
  };
  unsigned short* qkv_wT  = (unsigned short*)(ws + alloc(3072ull * 1024 * 2));
  unsigned short* proj_wT = (unsigned short*)(ws + alloc(1024ull * 1024 * 2));
  unsigned short* fc1_wT  = (unsigned short*)(ws + alloc(4096ull * 1024 * 2));
  unsigned short* fc2_wT  = (unsigned short*)(ws + alloc(1024ull * 4096 * 2));
  unsigned short* lnbuf   = (unsigned short*)(ws + alloc((size_t)TTOT * DIMC * 2));
  float*          hbuf    = (float*)(ws + alloc((size_t)TTOT * DIMC * 4));
  size_t qoff = alloc((size_t)TTOT * DIMC * 2);
  unsigned short* qb = (unsigned short*)(ws + qoff);
  unsigned short* kb = (unsigned short*)(ws + alloc((size_t)TTOT * DIMC * 2));
  unsigned short* vtb = (unsigned short*)(ws + alloc((VT1OFF + 8ull * 16 * 64 * NS1) * 2));
  unsigned short* aob = (unsigned short*)(ws + alloc((size_t)TTOT * DIMC * 2));
  // MLP activation aliases the (by-then consumed) q/k/vt/attn region (act is 16 KiB smaller).
  unsigned short* actb = (unsigned short*)(ws + qoff);

  // 1) weights -> bf16, transposed to [N,K]
  k_wt<<<(1024 * 3072 + 255) / 256, 256, 0, stream>>>(qkv_w,  qkv_wT,  1024, 3072);
  k_wt<<<(1024 * 1024 + 255) / 256, 256, 0, stream>>>(proj_w, proj_wT, 1024, 1024);
  k_wt<<<(1024 * 4096 + 255) / 256, 256, 0, stream>>>(fc1_w,  fc1_wT,  1024, 4096);
  k_wt<<<(4096 * 1024 + 255) / 256, 256, 0, stream>>>(fc2_w,  fc2_wT,  4096, 1024);

  // 2) LN1 over packed ragged input
  k_ln<<<TTOT / 8, 256, 0, stream>>>(x0, x1, TSPLIT, ln1_g, ln1_b, lnbuf);

  // 3) QKV GEMM  [7536,1024]x[1024,3072]  -> q, k, v^T (bf16)
  k_gemm_qkv<<<dim3(MTILES, 3072 / 512), 256, 0, stream>>>(lnbuf, qkv_wT, qkv_b,
                                                           qb, kb, vtb);
  // 4) flash attention: 7680 waves, 8 per block
  k_flash<<<(4 * 16 * QT0 + 8 * 16 * QT1) / 8, 256, 0, stream>>>(qb, kb, vtb, aob);

  // 5) proj + ls1 residual -> h (f32)
  k_gemm_proj<<<dim3(MTILES, 1024 / 512), 256, 0, stream>>>(aob, proj_wT, proj_b,
                                                            ls1, x0, x1, hbuf);
  // 6) LN2
  k_ln<<<TTOT / 8, 256, 0, stream>>>(hbuf, hbuf, TTOT, ln2_g, ln2_b, lnbuf);

  // 7) fc1 + GELU -> act (bf16), 8) fc2 + ls2 residual -> d_out (f32)
  k_gemm_fc1<<<dim3(MTILES, 4096 / 512), 256, 0, stream>>>(lnbuf, fc1_wT, fc1_b, actb);
  k_gemm_fc2<<<dim3(MTILES, 1024 / 512), 256, 0, stream>>>(actb, fc2_wT, fc2_b,
                                                           ls2, hbuf, (float*)d_out);
}
